// EncodeProcessDecode_16054587753003
// MI455X (gfx1250) — matrix operand order repaired and checked by reading, not compiled, not run on hardware
//
#include <hip/hip_runtime.h>
#include <hip/hip_bf16.h>
#include <math.h>

#define N_NODES 50000
#define N_EDGES 800000
#define NODE_FD 12
#define EDGE_FD 7
#define LATENT  128
#define OUT_F   3
#define MP_STEPS 5
#define WINDOW   4

typedef __bf16 bf16;
typedef __attribute__((ext_vector_type(16))) __bf16 bf16x16;
typedef __attribute__((ext_vector_type(8)))  __bf16 bf16x8;
typedef __attribute__((ext_vector_type(8)))  float  f32x8;

#define WMMA_BF16(a, b, c) \
  __builtin_amdgcn_wmma_f32_16x16x32_bf16(false, (a), false, (b), (short)0, (c), false, false)

// ---------------------------------------------------------------------------
// Fragment loaders (CDNA5 wave32 WMMA layouts, CDNA5 ISA §7.12.2)
// ---------------------------------------------------------------------------

// A fragment 16x32 bf16: this lane's row pointer; elements 0..7 <- K=k0+8h+0..7,
// elements 8..15 <- K=k0+16+8h+0..7  (h = lane>=16).
__device__ __forceinline__ bf16x16 a_frag_row(const bf16* rowp, int k0) {
  const int half = (threadIdx.x >> 4) & 1;
  const bf16x8 lo = *(const bf16x8*)(rowp + k0 + half * 8);
  const bf16x8 hi = *(const bf16x8*)(rowp + k0 + 16 + half * 8);
  bf16x16 a;
#pragma unroll
  for (int i = 0; i < 8; ++i) { a[i] = lo[i]; a[i + 8] = hi[i]; }
  return a;
}

// B fragment 32x16 bf16 from weights stored [Ncols][K] row-major:
// lane N = n0 + (l&15); elements 0..7 <- K=k0+8h+0..7, 8..15 <- K=k0+16+8h+0..7.
__device__ __forceinline__ bf16x16 b_frag(const bf16* Wt, int K, int n0, int k0) {
  const int n = n0 + (threadIdx.x & 15);
  const int half = (threadIdx.x >> 4) & 1;
  const bf16* p = Wt + (size_t)n * K + k0 + half * 8;
  const bf16x8 lo = *(const bf16x8*)(p);
  const bf16x8 hi = *(const bf16x8*)(p + 16);
  bf16x16 b;
#pragma unroll
  for (int i = 0; i < 8; ++i) { b[i] = lo[i]; b[i + 8] = hi[i]; }
  return b;
}

// C/D f32: element r -> row M = r + 8*(l>=16), col N = l&15 (within tile).
__device__ __forceinline__ void store_hidden(f32x8 c, const float* __restrict__ b1,
                                             bf16* hid, int HSTR) {
  const int n0 = (threadIdx.x >> 5) * 16;
  const int nn = n0 + (threadIdx.x & 15);
  const int half = (threadIdx.x >> 4) & 1;
  const float bias = b1[nn];
#pragma unroll
  for (int r = 0; r < 8; ++r) {
    const int m = r + 8 * half;
    hid[m * HSTR + nn] = (bf16)fmaxf(c[r] + bias, 0.f);
  }
}

// Second GEMM (K=128) + bias + ReLU into f32 LDS tile, then LN statistics.
__device__ __forceinline__ void gemm2_relu_ln(const bf16* hid, int HSTR,
                                              const bf16* __restrict__ W2t,
                                              const float* __restrict__ b2,
                                              float* outt, int OSTR,
                                              float* mrow, float* rrow) {
  const int n0 = (threadIdx.x >> 5) * 16;
  f32x8 c = {};
#pragma unroll
  for (int k0 = 0; k0 < LATENT; k0 += 32) {
    bf16x16 a = a_frag_row(hid + (threadIdx.x & 15) * HSTR, k0);
    bf16x16 b = b_frag(W2t, LATENT, n0, k0);
    c = WMMA_BF16(a, b, c);
  }
  const int nn = n0 + (threadIdx.x & 15);
  const int half = (threadIdx.x >> 4) & 1;
  const float bias = b2[nn];
#pragma unroll
  for (int r = 0; r < 8; ++r) {
    const int m = r + 8 * half;
    outt[m * OSTR + nn] = fmaxf(c[r] + bias, 0.f);
  }
  __syncthreads();
  if (threadIdx.x < 16) {
    const float* rp = outt + threadIdx.x * OSTR;
    float s = 0.f, s2 = 0.f;
#pragma unroll 4
    for (int k = 0; k < LATENT; ++k) { const float v = rp[k]; s += v; s2 += v * v; }
    const float mu  = s * (1.f / LATENT);
    const float var = s2 * (1.f / LATENT) - mu * mu;
    mrow[threadIdx.x] = mu;
    rrow[threadIdx.x] = rsqrtf(var + 1e-5f);
  }
  __syncthreads();
}

// ---------------------------------------------------------------------------
// Kernels
// ---------------------------------------------------------------------------

// Encoder MLP+LN from raw f32 features (K zero-padded to 32). 16 rows / block.
template <int DIN>
__global__ __launch_bounds__(256) void k_encoder(
    const float* __restrict__ src, const bf16* __restrict__ W1t,
    const float* __restrict__ b1, const bf16* __restrict__ W2t,
    const float* __restrict__ b2, const float* __restrict__ g,
    const float* __restrict__ beta, float* __restrict__ outf,
    bf16* __restrict__ out16) {
  __shared__ __align__(16) bf16  hid[16 * 136];
  __shared__ __align__(16) float outt[16 * 132];
  __shared__ float mrow[16], rrow[16];
  const int r0 = blockIdx.x * 16;
  const int row = r0 + (threadIdx.x & 15);
  const int half = (threadIdx.x >> 4) & 1;
  const float* rp = src + (size_t)row * DIN;
  bf16x16 a;
#pragma unroll
  for (int e = 0; e < 16; ++e) {
    const int k = ((e < 8) ? 0 : 16) + half * 8 + (e & 7);
    a[e] = (k < DIN) ? (bf16)rp[k] : (bf16)0.f;
  }
  const int n0 = (threadIdx.x >> 5) * 16;
  f32x8 c = {};
  c = WMMA_BF16(a, b_frag(W1t, 32, n0, 0), c);
  store_hidden(c, b1, hid, 136);
  __syncthreads();
  gemm2_relu_ln(hid, 136, W2t, b2, outt, 132, mrow, rrow);
  const int r = threadIdx.x >> 4, cb = threadIdx.x & 15;
  const float mu = mrow[r], rs = rrow[r];
  const size_t base = (size_t)(r0 + r) * LATENT;
#pragma unroll
  for (int j = 0; j < 8; ++j) {
    const int cc = cb + j * 16;
    const float v = (outt[r * 132 + cc] - mu) * rs * g[cc] + beta[cc];
    outf[base + cc] = v;
    out16[base + cc] = (bf16)v;
  }
}

// edge_net MLP+LN. MODE 0: message -> atomic scatter-add into aggr[idx1].
// MODE 1: edge update -> residual in-place on elf/el16out.
template <int MODE>
__global__ __launch_bounds__(256) void k_edge_mlp(
    const bf16* __restrict__ nl16, const bf16* __restrict__ el16,
    const int* __restrict__ idx1, const int* __restrict__ idx2,
    const bf16* __restrict__ W1t, const float* __restrict__ b1,
    const bf16* __restrict__ W2t, const float* __restrict__ b2,
    const float* __restrict__ g, const float* __restrict__ beta,
    float* __restrict__ aggr, float* __restrict__ elf,
    bf16* __restrict__ el16out) {
  __shared__ __align__(16) bf16  hid[16 * 136];
  __shared__ __align__(16) float outt[16 * 132];
  __shared__ float mrow[16], rrow[16];
  __shared__ int tgts[16];
  const int e0 = blockIdx.x * 16;
  const int e = e0 + (threadIdx.x & 15);
  const int i1 = idx1[e];
  const int i2 = idx2[e];
  if (MODE == 0 && threadIdx.x < 16) tgts[threadIdx.x] = idx1[e0 + threadIdx.x];
  const bf16* seg0 = nl16 + (size_t)i1 * LATENT;
  const bf16* seg1 = nl16 + (size_t)i2 * LATENT;
  const bf16* seg2 = el16 + (size_t)e * LATENT;
  const int n0 = (threadIdx.x >> 5) * 16;
  f32x8 c = {};
#pragma unroll
  for (int kc = 0; kc < 12; ++kc) {
    const int k0 = kc * 32;
    const bf16* rp = (k0 < 128) ? seg0 : ((k0 < 256) ? seg1 : seg2);
    c = WMMA_BF16(a_frag_row(rp, k0 & 127), b_frag(W1t, 384, n0, k0), c);
  }
  store_hidden(c, b1, hid, 136);
  __syncthreads();
  gemm2_relu_ln(hid, 136, W2t, b2, outt, 132, mrow, rrow);
  const int r = threadIdx.x >> 4, cb = threadIdx.x & 15;
  const float mu = mrow[r], rs = rrow[r];
#pragma unroll
  for (int j = 0; j < 8; ++j) {
    const int cc = cb + j * 16;
    const float v = (outt[r * 132 + cc] - mu) * rs * g[cc] + beta[cc];
    if (MODE == 0) {
      atomicAdd(&aggr[(size_t)tgts[r] * LATENT + cc], v);
    } else {
      const size_t o = (size_t)(e0 + r) * LATENT + cc;
      const float nv = v + elf[o];
      elf[o] = nv;
      el16out[o] = (bf16)nv;
    }
  }
}

// node_net MLP+LN with residual: input concat(aggr, nl), K=256.
__global__ __launch_bounds__(256) void k_node_net(
    const bf16* __restrict__ aggr16, const bf16* __restrict__ nl16,
    const float* __restrict__ nlf,
    const bf16* __restrict__ W1t, const float* __restrict__ b1,
    const bf16* __restrict__ W2t, const float* __restrict__ b2,
    const float* __restrict__ g, const float* __restrict__ beta,
    float* __restrict__ outf, bf16* __restrict__ out16) {
  __shared__ __align__(16) bf16  hid[16 * 136];
  __shared__ __align__(16) float outt[16 * 132];
  __shared__ float mrow[16], rrow[16];
  const int r0 = blockIdx.x * 16;
  const int row = r0 + (threadIdx.x & 15);
  const bf16* seg0 = aggr16 + (size_t)row * LATENT;
  const bf16* seg1 = nl16 + (size_t)row * LATENT;
  const int n0 = (threadIdx.x >> 5) * 16;
  f32x8 c = {};
#pragma unroll
  for (int kc = 0; kc < 8; ++kc) {
    const int k0 = kc * 32;
    const bf16* rp = (k0 < 128) ? seg0 : seg1;
    c = WMMA_BF16(a_frag_row(rp, k0 & 127), b_frag(W1t, 256, n0, k0), c);
  }
  store_hidden(c, b1, hid, 136);
  __syncthreads();
  gemm2_relu_ln(hid, 136, W2t, b2, outt, 132, mrow, rrow);
  const int r = threadIdx.x >> 4, cb = threadIdx.x & 15;
  const float mu = mrow[r], rs = rrow[r];
  const size_t base = (size_t)(r0 + r) * LATENT;
#pragma unroll
  for (int j = 0; j < 8; ++j) {
    const int cc = cb + j * 16;
    const float v = (outt[r * 132 + cc] - mu) * rs * g[cc] + beta[cc] + nlf[base + cc];
    outf[base + cc] = v;
    out16[base + cc] = (bf16)v;
  }
}

// GRU cell: gi = x@Wih.T + bih, gh = h@Whh.T + bhh (48 WMMA tiles), then gates.
__global__ __launch_bounds__(256) void k_gru(
    const bf16* __restrict__ x16, const bf16* __restrict__ h16,
    const float* __restrict__ hf,
    const bf16* __restrict__ Wih16, const bf16* __restrict__ Whh16,
    const float* __restrict__ bih, const float* __restrict__ bhh,
    float* __restrict__ outf, bf16* __restrict__ out16) {
  __shared__ __align__(16) float gbuf[16 * 776];  // [16][768+pad]: gi | gh
  const int r0 = blockIdx.x * 16;
  const int lrow = threadIdx.x & 15;
  const int half = (threadIdx.x >> 4) & 1;
  for (int tt = (threadIdx.x >> 5); tt < 48; tt += 8) {
    const bool isI = tt < 24;
    const int n0 = (isI ? tt : tt - 24) * 16;
    const bf16* A = (isI ? x16 : h16) + (size_t)(r0 + lrow) * LATENT;
    const bf16* B = isI ? Wih16 : Whh16;
    f32x8 c = {};
#pragma unroll
    for (int k0 = 0; k0 < LATENT; k0 += 32)
      c = WMMA_BF16(a_frag_row(A, k0), b_frag(B, LATENT, n0, k0), c);
    const int nn = n0 + (threadIdx.x & 15);
    const float bias = (isI ? bih : bhh)[nn];
    const int cbase = (isI ? 0 : 384) + nn;
#pragma unroll
    for (int r = 0; r < 8; ++r) gbuf[(r + 8 * half) * 776 + cbase] = c[r] + bias;
  }
  __syncthreads();
  const int r = threadIdx.x >> 4, cb = threadIdx.x & 15;
  const size_t node = (size_t)r0 + r;
#pragma unroll
  for (int j = 0; j < 8; ++j) {
    const int cc = cb + j * 16;
    const float gi_r = gbuf[r * 776 + cc];
    const float gi_z = gbuf[r * 776 + 128 + cc];
    const float gi_n = gbuf[r * 776 + 256 + cc];
    const float gh_r = gbuf[r * 776 + 384 + cc];
    const float gh_z = gbuf[r * 776 + 512 + cc];
    const float gh_n = gbuf[r * 776 + 640 + cc];
    const float rg = 1.f / (1.f + __expf(-(gi_r + gh_r)));
    const float zg = 1.f / (1.f + __expf(-(gi_z + gh_z)));
    const float ng = tanhf(gi_n + rg * gh_n);
    const float h = hf[node * LATENT + cc];
    const float hn = (1.f - zg) * ng + zg * h;
    outf[node * LATENT + cc] = hn;
    out16[node * LATENT + cc] = (bf16)hn;
  }
}

// Decoder: relu(h2@W1+b1)@W2+b2 -> [N,3].
__global__ __launch_bounds__(256) void k_dec(
    const bf16* __restrict__ h16, const bf16* __restrict__ W1t,
    const float* __restrict__ b1, const float* __restrict__ W2,
    const float* __restrict__ b2, float* __restrict__ out) {
  __shared__ __align__(16) float hb[16 * 132];
  const int r0 = blockIdx.x * 16;
  const int n0 = (threadIdx.x >> 5) * 16;
  const bf16* A = h16 + (size_t)(r0 + (threadIdx.x & 15)) * LATENT;
  f32x8 c = {};
#pragma unroll
  for (int k0 = 0; k0 < LATENT; k0 += 32)
    c = WMMA_BF16(a_frag_row(A, k0), b_frag(W1t, LATENT, n0, k0), c);
  const int nn = n0 + (threadIdx.x & 15);
  const int half = (threadIdx.x >> 4) & 1;
  const float bias = b1[nn];
#pragma unroll
  for (int r = 0; r < 8; ++r) hb[(r + 8 * half) * 132 + nn] = fmaxf(c[r] + bias, 0.f);
  __syncthreads();
  if (threadIdx.x < 48) {
    const int r = threadIdx.x / 3, o = threadIdx.x % 3;
    float s = b2[o];
#pragma unroll 4
    for (int k = 0; k < LATENT; ++k) s += hb[r * 132 + k] * W2[k * 3 + o];
    out[(size_t)(r0 + r) * 3 + o] = s;
  }
}

// Weight prep: transpose [din,dout] f32 -> [dout,Kpad] bf16 (zero-padded K).
__global__ void k_wt_transpose(const float* __restrict__ W, bf16* __restrict__ Wt,
                               int din, int dout, int Kpad) {
  const int i = blockIdx.x * blockDim.x + threadIdx.x;
  if (i >= dout * Kpad) return;
  const int n = i / Kpad, k = i % Kpad;
  Wt[i] = (k < din) ? (bf16)W[(size_t)k * dout + n] : (bf16)0.f;
}

__global__ void k_f32_to_bf16(const float* __restrict__ s, bf16* __restrict__ d, size_t n) {
  const size_t i = (size_t)blockIdx.x * blockDim.x + threadIdx.x;
  if (i < n) d[i] = (bf16)s[i];
}

// ---------------------------------------------------------------------------
// Host orchestration
// ---------------------------------------------------------------------------
extern "C" void kernel_launch(void* const* d_in, const int* in_sizes, int n_in,
                              void* d_out, int out_size, void* d_ws, size_t ws_size,
                              hipStream_t stream) {
  const float* x_seq  = (const float*)d_in[0];
  const float* ea_seq = (const float*)d_in[1];
  const int*   eidx   = (const int*)d_in[2];
  const int*   e_row  = eidx;
  const int*   e_col  = eidx + N_EDGES;
  auto P = [&](int i) { return (const float*)d_in[i]; };
  const float *ne_W1 = P(3),  *ne_b1 = P(4),  *ne_W2 = P(5),  *ne_b2 = P(6),  *ne_g = P(7),  *ne_be = P(8);
  const float *ee_W1 = P(9),  *ee_b1 = P(10), *ee_W2 = P(11), *ee_b2 = P(12), *ee_g = P(13), *ee_be = P(14);
  const float *en_W1 = P(15), *en_b1 = P(16), *en_W2 = P(17), *en_b2 = P(18), *en_g = P(19), *en_be = P(20);
  const float *nn_W1 = P(21), *nn_b1 = P(22), *nn_W2 = P(23), *nn_b2 = P(24), *nn_g = P(25), *nn_be = P(26);
  const float *g1_Wih = P(27), *g1_Whh = P(28), *g1_bih = P(29), *g1_bhh = P(30);
  const float *g2_Wih = P(31), *g2_Whh = P(32), *g2_bih = P(33), *g2_bhh = P(34);
  const float *dc_W1 = P(35), *dc_b1 = P(36), *dc_W2 = P(37), *dc_b2 = P(38);

  size_t off = 0;
  auto carve = [&](size_t bytes) -> void* {
    void* p = (char*)d_ws + off;
    off += (bytes + 255) & ~(size_t)255;
    return p;
  };
  // bf16 weight buffers ([N][K] layout)
  bf16* ne_W1t = (bf16*)carve((size_t)128 * 32 * 2);
  bf16* ee_W1t = (bf16*)carve((size_t)128 * 32 * 2);
  bf16* ne_W2t = (bf16*)carve((size_t)128 * 128 * 2);
  bf16* ee_W2t = (bf16*)carve((size_t)128 * 128 * 2);
  bf16* en_W1t = (bf16*)carve((size_t)128 * 384 * 2);
  bf16* en_W2t = (bf16*)carve((size_t)128 * 128 * 2);
  bf16* nn_W1t = (bf16*)carve((size_t)128 * 256 * 2);
  bf16* nn_W2t = (bf16*)carve((size_t)128 * 128 * 2);
  bf16* g1_Wih16 = (bf16*)carve((size_t)384 * 128 * 2);
  bf16* g1_Whh16 = (bf16*)carve((size_t)384 * 128 * 2);
  bf16* g2_Wih16 = (bf16*)carve((size_t)384 * 128 * 2);
  bf16* g2_Whh16 = (bf16*)carve((size_t)384 * 128 * 2);
  bf16* dc_W1t = (bf16*)carve((size_t)128 * 128 * 2);
  // activations
  const size_t NL = (size_t)N_NODES * LATENT, EL = (size_t)N_EDGES * LATENT;
  float* nlA    = (float*)carve(NL * 4);
  float* nlB    = (float*)carve(NL * 4);
  bf16*  nlA16  = (bf16*)carve(NL * 2);
  bf16*  nlB16  = (bf16*)carve(NL * 2);
  float* elf    = (float*)carve(EL * 4);
  bf16*  el16   = (bf16*)carve(EL * 2);
  float* aggr   = (float*)carve(NL * 4);
  bf16*  aggr16 = (bf16*)carve(NL * 2);
  float* h1f    = (float*)carve(NL * 4);
  float* h2f    = (float*)carve(NL * 4);
  bf16*  h1b    = (bf16*)carve(NL * 2);
  bf16*  h2b    = (bf16*)carve(NL * 2);

  // --- weight prep (every call: deterministic, no caching) ---
  auto tgrid = [](int n) { return dim3((n + 255) / 256); };
  k_wt_transpose<<<tgrid(128 * 32), 256, 0, stream>>>(ne_W1, ne_W1t, NODE_FD, 128, 32);
  k_wt_transpose<<<tgrid(128 * 32), 256, 0, stream>>>(ee_W1, ee_W1t, EDGE_FD, 128, 32);
  k_wt_transpose<<<tgrid(128 * 128), 256, 0, stream>>>(ne_W2, ne_W2t, 128, 128, 128);
  k_wt_transpose<<<tgrid(128 * 128), 256, 0, stream>>>(ee_W2, ee_W2t, 128, 128, 128);
  k_wt_transpose<<<tgrid(128 * 384), 256, 0, stream>>>(en_W1, en_W1t, 384, 128, 384);
  k_wt_transpose<<<tgrid(128 * 128), 256, 0, stream>>>(en_W2, en_W2t, 128, 128, 128);
  k_wt_transpose<<<tgrid(128 * 256), 256, 0, stream>>>(nn_W1, nn_W1t, 256, 128, 256);
  k_wt_transpose<<<tgrid(128 * 128), 256, 0, stream>>>(nn_W2, nn_W2t, 128, 128, 128);
  k_wt_transpose<<<tgrid(128 * 128), 256, 0, stream>>>(dc_W1, dc_W1t, 128, 128, 128);
  // GRU: x@Wih.T needs B in [N=3L][K=L] layout == Wih row-major, straight convert.
  k_f32_to_bf16<<<tgrid(384 * 128), 256, 0, stream>>>(g1_Wih, g1_Wih16, 384 * 128);
  k_f32_to_bf16<<<tgrid(384 * 128), 256, 0, stream>>>(g1_Whh, g1_Whh16, 384 * 128);
  k_f32_to_bf16<<<tgrid(384 * 128), 256, 0, stream>>>(g2_Wih, g2_Wih16, 384 * 128);
  k_f32_to_bf16<<<tgrid(384 * 128), 256, 0, stream>>>(g2_Whh, g2_Whh16, 384 * 128);

  // h0 = zeros
  hipMemsetAsync(h1f, 0, NL * 4, stream);
  hipMemsetAsync(h2f, 0, NL * 4, stream);
  hipMemsetAsync(h1b, 0, NL * 2, stream);
  hipMemsetAsync(h2b, 0, NL * 2, stream);

  const dim3 NB(N_NODES / 16), EB(N_EDGES / 16);
  for (int t = 0; t < WINDOW; ++t) {
    k_encoder<NODE_FD><<<NB, 256, 0, stream>>>(x_seq + (size_t)t * N_NODES * NODE_FD,
        ne_W1t, ne_b1, ne_W2t, ne_b2, ne_g, ne_be, nlA, nlA16);
    k_encoder<EDGE_FD><<<EB, 256, 0, stream>>>(ea_seq + (size_t)t * N_EDGES * EDGE_FD,
        ee_W1t, ee_b1, ee_W2t, ee_b2, ee_g, ee_be, elf, el16);
    for (int s = 0; s < MP_STEPS; ++s) {
      hipMemsetAsync(aggr, 0, NL * 4, stream);
      // message: cat(nl[col], nl[row], el) -> scatter-add at col
      k_edge_mlp<0><<<EB, 256, 0, stream>>>(nlA16, el16, e_col, e_row,
          en_W1t, en_b1, en_W2t, en_b2, en_g, en_be, aggr, nullptr, nullptr);
      k_f32_to_bf16<<<tgrid((int)NL), 256, 0, stream>>>(aggr, aggr16, NL);
      // node update: cat(aggr, nl) + residual -> nlB
      k_node_net<<<NB, 256, 0, stream>>>(aggr16, nlA16, nlA,
          nn_W1t, nn_b1, nn_W2t, nn_b2, nn_g, nn_be, nlB, nlB16);
      // edge update: cat(nl[row], nl[col], el) + residual, in-place on el
      k_edge_mlp<1><<<EB, 256, 0, stream>>>(nlA16, el16, e_row, e_col,
          en_W1t, en_b1, en_W2t, en_b2, en_g, en_be, nullptr, elf, el16);
      float* tf = nlA; nlA = nlB; nlB = tf;
      bf16* tb = nlA16; nlA16 = nlB16; nlB16 = tb;
    }
    k_gru<<<NB, 256, 0, stream>>>(nlA16, h1b, h1f, g1_Wih16, g1_Whh16, g1_bih, g1_bhh, h1f, h1b);
    k_gru<<<NB, 256, 0, stream>>>(h1b, h2b, h2f, g2_Wih16, g2_Whh16, g2_bih, g2_bhh, h2f, h2b);
    k_dec<<<NB, 256, 0, stream>>>(h2b, dc_W1t, dc_b1, dc_W2, dc_b2,
                                  (float*)d_out + (size_t)t * N_NODES * OUT_F);
  }
  (void)in_sizes; (void)n_in; (void)out_size; (void)ws_size;
}
